// rnn_decoder_3899830305359
// MI455X (gfx1250) — matrix operand order repaired and checked by reading, not compiled, not run on hardware
//
#include <hip/hip_runtime.h>

// ---------------------------------------------------------------------------
// MI455X (gfx1250) dual-branch attentional LSTM decoder.
// All GEMMs run on v_wmma_f32_16x16x32_bf16 (fp32 accumulate).
//  * Weights: packed fp32->bf16 once per launch into the exact WMMA
//    B-fragment layout (lane = N col + 16*((k>>3)&1), 16 contiguous bf16 = 16 of the block's 32 K).
//  * Activations: packed fp32->bf16 once per GEMM into the exact WMMA
//    A-fragment layout, so the GEMM inner loop is two coalesced 32B fragment
//    loads + one WMMA -- no f32->bf16 conversion in the hot loop, no tails.
//  * ~175MB of bf16 weights stay mostly L2-resident across the 50 steps.
//  * K=30000 soft@emb GEMM uses split-K (grid.z=8) with a deterministic
//    reduction (no float atomics -> bit-identical across graph replays).
// ---------------------------------------------------------------------------

#define EMBD 512
#define HIDD 1024
#define BATCH 32
#define TSTEPS 50
#define SLEN 128
#define VOC 30000
#define DVOC 5000
#define NSPLIT 8

typedef __attribute__((ext_vector_type(16))) __bf16 v16bf;
typedef __attribute__((ext_vector_type(8)))  float  v8f;

// ---------------------------------------------------------------- helpers ---
__device__ __forceinline__ unsigned short f2bf(float f) {
  unsigned u = __float_as_uint(f);
  u += 0x7fffu + ((u >> 16) & 1u);            // round-to-nearest-even
  return (unsigned short)(u >> 16);
}
__device__ __forceinline__ float sigm(float x) {
  return 1.0f / (1.0f + __expf(-x));
}

// ------------------------------------------------------------ pack kernels ---
// W [N,K] row-major fp32 -> Wp bf16 in WMMA-B layout: for tile tn=n/16, kk=k%32,
// Wp[(tn*Kp + k-kk + ((kk>>3)&1)*16 + n%16)*16 + (kk&7) + ((kk>>4)&1)*8] = W[n][k].  Zero-padded to Np, Kp.
__global__ void pack_w(const float* __restrict__ W, unsigned short* __restrict__ Wp,
                       int N, int K, int Np, int Kp) {
  size_t tot = (size_t)Np * Kp;
  size_t idx = (size_t)blockIdx.x * blockDim.x + threadIdx.x;
  if (idx >= tot) return;
  int k = (int)(idx % Kp);
  int n = (int)(idx / Kp);
  float v = (n < N && k < K) ? W[(size_t)n * K + k] : 0.0f;
  Wp[((size_t)(n >> 4) * Kp + (k & ~31) + ((k >> 3) & 1) * 16 + (n & 15)) * 16 + (k & 7) + ((k >> 4) & 1) * 8] = f2bf(v);
}

// Bm [K,N] row-major fp32 (a K x N "B" matrix, e.g. the embedding table for
// soft@emb) -> same packed layout.
__global__ void pack_b(const float* __restrict__ Bm, unsigned short* __restrict__ Bp,
                       int K, int N, int Np, int Kp) {
  size_t tot = (size_t)Np * Kp;
  size_t idx = (size_t)blockIdx.x * blockDim.x + threadIdx.x;
  if (idx >= tot) return;
  int k = (int)(idx % Kp);
  int n = (int)(idx / Kp);
  float v = (n < N && k < K) ? Bm[(size_t)k * N + n] : 0.0f;
  Bp[((size_t)(n >> 4) * Kp + (k & ~31) + ((k >> 3) & 1) * 16 + (n & 15)) * 16 + (k & 7) + ((k >> 4) & 1) * 8] = f2bf(v);
}

// X [32,K] fp32 activations -> Xa bf16 in WMMA-A fragment layout.
// A fragment (ISA 7.12.2, 16-bit A 16x32): lane holds row m=(lane&15)+16*mt;
// halves j<8 -> k=kb*32+g*8+j, halves j>=8 -> k=kb*32+16+g*8+(j-8), g=lane>>4.
// Storage: Xa[(((mt*Kb)+kb)*32 + lane)*16 + j]  (lane-coalesced 32B chunks).
__global__ void pack_a(const float* __restrict__ X, unsigned short* __restrict__ Xa,
                       int K, int Kb) {
  int idx = blockIdx.x * 256 + threadIdx.x;
  int tot = 1024 * Kb;                        // 2 mtiles * Kb * 32 lanes * 16
  if (idx >= tot) return;
  int j    = idx & 15;
  int lane = (idx >> 4) & 31;
  int tmp  = idx >> 9;                        // mt*Kb + kb
  int kb   = tmp % Kb;
  int mt   = tmp / Kb;
  int g    = lane >> 4;
  int m    = mt * 16 + (lane & 15);
  int k    = kb * 32 + ((j < 8) ? (g * 8 + j) : (16 + g * 8 + (j - 8)));
  float v  = (k < K) ? X[(size_t)m * K + k] : 0.0f;
  Xa[idx] = f2bf(v);
}

// ------------------------------------------------------------- WMMA GEMM ----
#define WMMA_BF16(A, B, C) \
  __builtin_amdgcn_wmma_f32_16x16x32_bf16(false, (A), false, (B), (short)0, (C), false, false)

// Y[32,N] = A1 @ B1 (+ A2 @ B2) + bias, optional tanh.  A/B pre-packed bf16.
// grid = (Np/16, 2), block = 32 (one wave per 16x16 tile, EXEC all ones).
// Inner loop: one 32B A-fragment load + one 32B B-fragment load + one WMMA.
__global__ void gemm_wmma(const unsigned short* __restrict__ A1,
                          const unsigned short* __restrict__ B1, int K1b,
                          const unsigned short* __restrict__ A2,
                          const unsigned short* __restrict__ B2, int K2b,
                          const float* __restrict__ bias,
                          float* __restrict__ Y, int N, int act) {
  int tn   = blockIdx.x;
  int mt   = blockIdx.y;          // 0 or 1 -> rows 0..15 / 16..31
  int lane = threadIdx.x;         // 0..31, wave32

  v8f acc = {0.f, 0.f, 0.f, 0.f, 0.f, 0.f, 0.f, 0.f};

  {
    const unsigned short* ab = A1 + (size_t)mt * K1b * 512 + lane * 16;
    const unsigned short* bb = B1 + (size_t)tn * K1b * 512 + lane * 16;
    for (int kb = 0; kb < K1b; ++kb) {
      v16bf a = *(const v16bf*)(ab + (size_t)kb * 512);
      v16bf b = *(const v16bf*)(bb + (size_t)kb * 512);
      if (kb + 2 < K1b)
        __builtin_prefetch((const void*)(bb + (size_t)(kb + 2) * 512), 0, 1);
      acc = WMMA_BF16(a, b, acc);
    }
  }
  if (A2) {
    const unsigned short* ab = A2 + (size_t)mt * K2b * 512 + lane * 16;
    const unsigned short* bb = B2 + (size_t)tn * K2b * 512 + lane * 16;
    for (int kb = 0; kb < K2b; ++kb) {
      v16bf a = *(const v16bf*)(ab + (size_t)kb * 512);
      v16bf b = *(const v16bf*)(bb + (size_t)kb * 512);
      acc = WMMA_BF16(a, b, acc);
    }
  }

  // D layout: lane holds col n=lane&15; VGPR r -> row r + 8*(lane>>4).
  int g   = lane >> 4;
  int col = tn * 16 + (lane & 15);
  if (col < N) {
    float bv = bias ? bias[col] : 0.0f;
#pragma unroll
    for (int r = 0; r < 8; ++r) {
      int row = mt * 16 + g * 8 + r;
      float v = acc[r] + bv;
      if (act == 1) v = tanhf(v);
      Y[(size_t)row * N + col] = v;
    }
  }
}

// Split-K variant: grid = (Np/16, 2, NSPLIT); slice z covers K blocks
// [z*chunkb, z*chunkb+chunkb); partials to P[(z*32+row)*N+col].
__global__ void gemm_wmma_sk(const unsigned short* __restrict__ A,
                             const unsigned short* __restrict__ B, int Kb,
                             float* __restrict__ P, int N, int chunkb) {
  int tn   = blockIdx.x;
  int mt   = blockIdx.y;
  int z    = blockIdx.z;
  int lane = threadIdx.x;

  int kbeg = z * chunkb;
  int kend = kbeg + chunkb; if (kend > Kb) kend = Kb;
  v8f acc = {0.f, 0.f, 0.f, 0.f, 0.f, 0.f, 0.f, 0.f};

  const unsigned short* ab = A + (size_t)mt * Kb * 512 + lane * 16;
  const unsigned short* bb = B + (size_t)tn * Kb * 512 + lane * 16;
  for (int kb = kbeg; kb < kend; ++kb) {
    v16bf a = *(const v16bf*)(ab + (size_t)kb * 512);
    v16bf b = *(const v16bf*)(bb + (size_t)kb * 512);
    if (kb + 2 < kend)
      __builtin_prefetch((const void*)(bb + (size_t)(kb + 2) * 512), 0, 1);
    acc = WMMA_BF16(a, b, acc);
  }

  int g   = lane >> 4;
  int col = tn * 16 + (lane & 15);
  if (col < N) {
#pragma unroll
    for (int r = 0; r < 8; ++r) {
      int rw = mt * 16 + g * 8 + r;
      P[((size_t)z * 32 + rw) * N + col] = acc[r];
    }
  }
}

// Deterministic split-K reduction: Y[i] = sum_z P[z*stride + i].
__global__ void reduce_sk_k(const float* __restrict__ P, float* __restrict__ Y,
                            int stride, int n) {
  int i = blockIdx.x * 256 + threadIdx.x;
  if (i >= n) return;
  float s = 0.f;
#pragma unroll
  for (int z = 0; z < NSPLIT; ++z) s += P[(size_t)z * stride + i];
  Y[i] = s;
}

// --------------------------------------------------------- pointwise etc. ---
__global__ void lookup_k(const int* __restrict__ ids, const float* __restrict__ tab,
                         float* __restrict__ out) {
  int i = blockIdx.x * 256 + threadIdx.x;
  if (i >= BATCH * EMBD) return;
  int b = i >> 9, e = i & (EMBD - 1);
  out[i] = tab[(size_t)ids[b] * EMBD + e];
}

__global__ void copy2_k(const float* __restrict__ s, float* __restrict__ a,
                        float* __restrict__ b, int n) {
  int i = blockIdx.x * 256 + threadIdx.x;
  if (i < n) { a[i] = s[i]; b[i] = s[i]; }
}

__global__ void add2_k(const float* __restrict__ a, const float* __restrict__ b,
                       float* __restrict__ o, int n) {
  int i = blockIdx.x * 256 + threadIdx.x;
  if (i < n) o[i] = a[i] + b[i];
}

__global__ void gate_mix_k(const float* __restrict__ y1, const float* __restrict__ y2,
                           const float* __restrict__ e1, const float* __restrict__ e2,
                           float* __restrict__ out, int n) {
  int i = blockIdx.x * 256 + threadIdx.x;
  if (i >= n) return;
  float gm = sigm(y1[i] + y2[i]);
  out[i] = gm * e1[i] + (1.0f - gm) * e2[i];
}

// gates [32,4096] in i,f,g,o order; in-place h/c update.
__global__ void lstm_elem_k(const float* __restrict__ gates,
                            float* __restrict__ h, float* __restrict__ c) {
  int idx = blockIdx.x * 256 + threadIdx.x;
  if (idx >= BATCH * HIDD) return;
  int b = idx / HIDD, j = idx % HIDD;
  const float* gr = gates + (size_t)b * 4 * HIDD;
  float ig = sigm(gr[j]);
  float fg = sigm(gr[HIDD + j]);
  float gg = tanhf(gr[2 * HIDD + j]);
  float og = sigm(gr[3 * HIDD + j]);
  float cn = fg * c[idx] + ig * gg;
  c[idx] = cn;
  h[idx] = og * tanhf(cn);
}

// scores[b,s] = dot(contexts[b,s,:], gamma[b,:]) over HIDD; grid = B*S.
__global__ void attn_scores_k(const float* __restrict__ ctxs,
                              const float* __restrict__ gamma,
                              float* __restrict__ sc) {
  int bs = blockIdx.x;
  int b = bs >> 7, s = bs & (SLEN - 1);
  const float* cv = ctxs + ((size_t)b * SLEN + s) * HIDD;
  const float* gv = gamma + (size_t)b * HIDD;
  float sum = 0.f;
  for (int i = threadIdx.x; i < HIDD; i += 256) sum += cv[i] * gv[i];
  __shared__ float red[256];
  red[threadIdx.x] = sum; __syncthreads();
  for (int o = 128; o > 0; o >>= 1) {
    if (threadIdx.x < o) red[threadIdx.x] += red[threadIdx.x + o];
    __syncthreads();
  }
  if (threadIdx.x == 0) sc[b * SLEN + s] = red[0];
}

// ctx[b,h] = sum_s w[b,s]*contexts[b,s,h]; grid = B.
__global__ void attn_ctx_k(const float* __restrict__ w,
                           const float* __restrict__ ctxs,
                           float* __restrict__ out) {
  int b = blockIdx.x;
  for (int h = threadIdx.x; h < HIDD; h += 256) {
    float sum = 0.f;
    for (int s = 0; s < SLEN; ++s)
      sum += w[b * SLEN + s] * ctxs[((size_t)b * SLEN + s) * HIDD + h];
    out[b * HIDD + h] = sum;
  }
}

__global__ void concat2_k(const float* __restrict__ ctx, const float* __restrict__ h,
                          float* __restrict__ cat) {
  int i = blockIdx.x * 256 + threadIdx.x;
  if (i >= BATCH * 2 * HIDD) return;
  int b = i / (2 * HIDD), j = i % (2 * HIDD);
  cat[i] = (j < HIDD) ? ctx[b * HIDD + j] : h[b * HIDD + j - HIDD];
}

// numerically stable row softmax, rows = 32 (grid), block = 256.
__global__ void softmax_rows_k(const float* __restrict__ in,
                               float* __restrict__ out, int N) {
  int row = blockIdx.x, tid = threadIdx.x;
  const float* x = in + (size_t)row * N;
  float* y = out + (size_t)row * N;
  __shared__ float red[256];
  float mx = -3.4e38f;
  for (int i = tid; i < N; i += 256) mx = fmaxf(mx, x[i]);
  red[tid] = mx; __syncthreads();
  for (int o = 128; o > 0; o >>= 1) {
    if (tid < o) red[tid] = fmaxf(red[tid], red[tid + o]);
    __syncthreads();
  }
  mx = red[0]; __syncthreads();
  float s = 0.f;
  for (int i = tid; i < N; i += 256) s += __expf(x[i] - mx);
  red[tid] = s; __syncthreads();
  for (int o = 128; o > 0; o >>= 1) {
    if (tid < o) red[tid] += red[tid + o];
    __syncthreads();
  }
  float inv = 1.0f / red[0];
  for (int i = tid; i < N; i += 256) y[i] = __expf(x[i] - mx) * inv;
}

// ------------------------------------------------------------------- host ---
static inline int kp32(int x) { return (x + 31) & ~31; }
static inline int np16(int x) { return (x + 15) & ~15; }
static inline unsigned gblk(size_t n) { return (unsigned)((n + 255) / 256); }

struct BranchP {
  const int* ids; const float* tab; int V;
  unsigned short *embP, *wih0, *whh0, *wih1, *whh1;
  unsigned short *g1P, *g2P, *attinP, *attoutP, *linP;
  const float *g1b, *g2b, *attinb, *attoutb, *linb;
  float *bsum0, *bsum1, *soft;
  float *h0b, *c0b, *h1b, *c1b;
  float *outBase, *embx;
};

extern "C" void kernel_launch(void* const* d_in, const int* in_sizes, int n_in,
                              void* d_out, int out_size, void* d_ws, size_t ws_size,
                              hipStream_t stream) {
  (void)in_sizes; (void)n_in; (void)out_size; (void)ws_size;

  // ---- inputs (setup_inputs dict order, params flattened in insertion order)
  const int*   inputs   = (const int*)d_in[0];
  const int*   inputsD  = (const int*)d_in[1];
  const float* contexts = (const float*)d_in[2];
  const float* h0       = (const float*)d_in[3];
  const float* c0       = (const float*)d_in[4];
  const float* embT     = (const float*)d_in[5];
  const float* embdT    = (const float*)d_in[6];
  const float* Wih0  = (const float*)d_in[7];  const float* Whh0 = (const float*)d_in[8];
  const float* bih0  = (const float*)d_in[9];  const float* bhh0 = (const float*)d_in[10];
  const float* Wih1  = (const float*)d_in[11]; const float* Whh1 = (const float*)d_in[12];
  const float* bih1  = (const float*)d_in[13]; const float* bhh1 = (const float*)d_in[14];
  const float* Wih0d = (const float*)d_in[15]; const float* Whh0d = (const float*)d_in[16];
  const float* bih0d = (const float*)d_in[17]; const float* bhh0d = (const float*)d_in[18];
  const float* Wih1d = (const float*)d_in[19]; const float* Whh1d = (const float*)d_in[20];
  const float* bih1d = (const float*)d_in[21]; const float* bhh1d = (const float*)d_in[22];
  const float* attinW  = (const float*)d_in[23]; const float* attinb  = (const float*)d_in[24];
  const float* attoutW = (const float*)d_in[25]; const float* attoutb = (const float*)d_in[26];
  const float* attindW = (const float*)d_in[27]; const float* attindb = (const float*)d_in[28];
  const float* attoutdW= (const float*)d_in[29]; const float* attoutdb= (const float*)d_in[30];
  const float* linW  = (const float*)d_in[31]; const float* linb  = (const float*)d_in[32];
  const float* lindW = (const float*)d_in[33]; const float* lindb = (const float*)d_in[34];
  const float* g1W = (const float*)d_in[35]; const float* g1b = (const float*)d_in[36];
  const float* g2W = (const float*)d_in[37]; const float* g2b = (const float*)d_in[38];
  const float* g1dW = (const float*)d_in[39]; const float* g1db = (const float*)d_in[40];
  const float* g2dW = (const float*)d_in[41]; const float* g2db = (const float*)d_in[42];

  float* outS = (float*)d_out;                                   // [T,B,H]
  float* outD = outS + (size_t)TSTEPS * BATCH * HIDD;            // [T,B,H]
  float* gemb = outD + (size_t)TSTEPS * BATCH * HIDD;            // [T-1,B,E]

  // ---- workspace bump allocator -------------------------------------------
  char* cur = (char*)d_ws;
  auto allocH = [&](size_t n) -> unsigned short* {
    unsigned short* r = (unsigned short*)cur;
    cur += ((n * 2 + 255) & ~(size_t)255); return r;
  };
  auto allocF = [&](size_t n) -> float* {
    float* r = (float*)cur;
    cur += ((n * 4 + 255) & ~(size_t)255); return r;
  };

  const int VOCp = kp32(VOC), DVOCp = kp32(DVOC);
  unsigned short* embP   = allocH((size_t)EMBD * VOCp);
  unsigned short* embdP  = allocH((size_t)EMBD * DVOCp);
  unsigned short* wih0P  = allocH((size_t)4 * HIDD * EMBD);
  unsigned short* whh0P  = allocH((size_t)4 * HIDD * HIDD);
  unsigned short* wih1P  = allocH((size_t)4 * HIDD * HIDD);
  unsigned short* whh1P  = allocH((size_t)4 * HIDD * HIDD);
  unsigned short* wih0dP = allocH((size_t)4 * HIDD * EMBD);
  unsigned short* whh0dP = allocH((size_t)4 * HIDD * HIDD);
  unsigned short* wih1dP = allocH((size_t)4 * HIDD * HIDD);
  unsigned short* whh1dP = allocH((size_t)4 * HIDD * HIDD);
  unsigned short* attinP  = allocH((size_t)HIDD * HIDD);
  unsigned short* attindP = allocH((size_t)HIDD * HIDD);
  unsigned short* attoutP = allocH((size_t)HIDD * 2 * HIDD);
  unsigned short* attoutdP= allocH((size_t)HIDD * 2 * HIDD);
  unsigned short* linP  = allocH((size_t)np16(VOC) * HIDD);
  unsigned short* lindP = allocH((size_t)np16(DVOC) * HIDD);
  unsigned short* g1P  = allocH((size_t)EMBD * EMBD);
  unsigned short* g2P  = allocH((size_t)EMBD * EMBD);
  unsigned short* g1dP = allocH((size_t)EMBD * EMBD);
  unsigned short* g2dP = allocH((size_t)EMBD * EMBD);

  // packed A-fragment buffers (bf16), 32*Kp halfs each
  unsigned short* softA = allocH((size_t)32 * VOCp);
  unsigned short* emb1A = allocH(32 * EMBD);
  unsigned short* emb2A = allocH(32 * EMBD);
  unsigned short* embxA = allocH(32 * EMBD);
  unsigned short* h0A   = allocH(32 * HIDD);
  unsigned short* h1A   = allocH(32 * HIDD);
  unsigned short* catA  = allocH(32 * 2 * HIDD);
  unsigned short* outA  = allocH(32 * HIDD);

  float* bsum0  = allocF(4 * HIDD);  float* bsum1  = allocF(4 * HIDD);
  float* bsum0d = allocF(4 * HIDD);  float* bsum1d = allocF(4 * HIDD);
  float* soft   = allocF((size_t)BATCH * VOC);
  float* softd  = allocF((size_t)BATCH * DVOC);
  float* logits = allocF((size_t)BATCH * VOC);
  float* part   = allocF((size_t)NSPLIT * BATCH * EMBD);   // split-K partials
  float* hs0 = allocF(BATCH * HIDD); float* cs0 = allocF(BATCH * HIDD);
  float* hs1 = allocF(BATCH * HIDD); float* cs1 = allocF(BATCH * HIDD);
  float* hd0 = allocF(BATCH * HIDD); float* cd0 = allocF(BATCH * HIDD);
  float* hd1 = allocF(BATCH * HIDD); float* cd1 = allocF(BATCH * HIDD);
  float* gates = allocF((size_t)BATCH * 4 * HIDD);
  float* emb1 = allocF(BATCH * EMBD); float* emb2 = allocF(BATCH * EMBD);
  float* y1 = allocF(BATCH * EMBD);   float* y2 = allocF(BATCH * EMBD);
  float* embx  = allocF(BATCH * EMBD); float* embxd = allocF(BATCH * EMBD);
  float* gamma = allocF(BATCH * HIDD);
  float* sc  = allocF(BATCH * SLEN);  float* wts = allocF(BATCH * SLEN);
  float* ctxb = allocF(BATCH * HIDD); float* cat = allocF(BATCH * 2 * HIDD);

  // ---- launch helpers ------------------------------------------------------
  auto packW = [&](const float* W, unsigned short* Wp, int N, int K) {
    int Np = np16(N), Kp = kp32(K);
    size_t tot = (size_t)Np * Kp;
    pack_w<<<gblk(tot), 256, 0, stream>>>(W, Wp, N, K, Np, Kp);
  };
  auto packBm = [&](const float* Bm, unsigned short* Bp, int K, int N) {
    int Np = np16(N), Kp = kp32(K);
    size_t tot = (size_t)Np * Kp;
    pack_b<<<gblk(tot), 256, 0, stream>>>(Bm, Bp, K, N, Np, Kp);
  };
  auto packA = [&](const float* X, unsigned short* Xa, int K) {
    int Kb = kp32(K) / 32;
    pack_a<<<gblk((size_t)1024 * Kb), 256, 0, stream>>>(X, Xa, K, Kb);
  };
  auto gemm = [&](const unsigned short* A1, const unsigned short* B1, int K1,
                  const unsigned short* A2, const unsigned short* B2, int K2,
                  const float* bias, float* Y, int N, int act) {
    gemm_wmma<<<dim3(np16(N) / 16, 2), 32, 0, stream>>>(
        A1, B1, kp32(K1) / 32, A2, B2, kp32(K2) / 32, bias, Y, N, act);
  };
  // split-K GEMM: Y[32,N] = A @ Bp (no bias/act); used for soft@emb.
  auto gemmSK = [&](const unsigned short* A, const unsigned short* Bp, int K,
                    float* Y, int N) {
    int Kb = kp32(K) / 32;
    int chunkb = (Kb + NSPLIT - 1) / NSPLIT;
    gemm_wmma_sk<<<dim3(np16(N) / 16, 2, NSPLIT), 32, 0, stream>>>(
        A, Bp, Kb, part, N, chunkb);
    reduce_sk_k<<<gblk((size_t)BATCH * N), 256, 0, stream>>>(
        part, Y, BATCH * N, BATCH * N);
  };

  // ---- prep: pack all weights to bf16 B-fragment layout --------------------
  packBm(embT,  embP,  VOC,  EMBD);
  packBm(embdT, embdP, DVOC, EMBD);
  packW(Wih0,  wih0P,  4 * HIDD, EMBD);  packW(Whh0,  whh0P,  4 * HIDD, HIDD);
  packW(Wih1,  wih1P,  4 * HIDD, HIDD);  packW(Whh1,  whh1P,  4 * HIDD, HIDD);
  packW(Wih0d, wih0dP, 4 * HIDD, EMBD);  packW(Whh0d, whh0dP, 4 * HIDD, HIDD);
  packW(Wih1d, wih1dP, 4 * HIDD, HIDD);  packW(Whh1d, whh1dP, 4 * HIDD, HIDD);
  packW(attinW,  attinP,  HIDD, HIDD);   packW(attindW, attindP, HIDD, HIDD);
  packW(attoutW, attoutP, HIDD, 2 * HIDD); packW(attoutdW, attoutdP, HIDD, 2 * HIDD);
  packW(linW,  linP,  VOC,  HIDD);       packW(lindW, lindP, DVOC, HIDD);
  packW(g1W, g1P, EMBD, EMBD);  packW(g2W, g2P, EMBD, EMBD);
  packW(g1dW, g1dP, EMBD, EMBD); packW(g2dW, g2dP, EMBD, EMBD);

  add2_k<<<gblk(4 * HIDD), 256, 0, stream>>>(bih0, bhh0, bsum0, 4 * HIDD);
  add2_k<<<gblk(4 * HIDD), 256, 0, stream>>>(bih1, bhh1, bsum1, 4 * HIDD);
  add2_k<<<gblk(4 * HIDD), 256, 0, stream>>>(bih0d, bhh0d, bsum0d, 4 * HIDD);
  add2_k<<<gblk(4 * HIDD), 256, 0, stream>>>(bih1d, bhh1d, bsum1d, 4 * HIDD);

  // ---- initial state (both branches start from the same h0/c0) -------------
  int nBH = BATCH * HIDD;
  copy2_k<<<gblk(nBH), 256, 0, stream>>>(h0,        hs0, hd0, nBH);
  copy2_k<<<gblk(nBH), 256, 0, stream>>>(h0 + nBH,  hs1, hd1, nBH);
  copy2_k<<<gblk(nBH), 256, 0, stream>>>(c0,        cs0, cd0, nBH);
  copy2_k<<<gblk(nBH), 256, 0, stream>>>(c0 + nBH,  cs1, cd1, nBH);

  BranchP spk = { inputs,  embT,  VOC,
                  embP,  wih0P,  whh0P,  wih1P,  whh1P,
                  g1P,  g2P,  attinP,  attoutP,  linP,
                  g1b,  g2b,  attinb,  attoutb,  linb,
                  bsum0,  bsum1,  soft,
                  hs0, cs0, hs1, cs1, outS, embx };
  BranchP dir = { inputsD, embdT, DVOC,
                  embdP, wih0dP, whh0dP, wih1dP, whh1dP,
                  g1dP, g2dP, attindP, attoutdP, lindP,
                  g1db, g2db, attindb, attoutdb, lindb,
                  bsum0d, bsum1d, softd,
                  hd0, cd0, hd1, cd1, outD, embxd };

  auto runStep = [&](const BranchP& R, int t) {
    const int* idt = R.ids + t * BATCH;
    const unsigned short* xinA;
    lookup_k<<<gblk(BATCH * EMBD), 256, 0, stream>>>(idt, R.tab, emb2);
    packA(emb2, emb2A, EMBD);
    if (t == 0) {
      xinA = emb2A;
    } else {
      // emb1 = soft_prev @ E  (K = V huge, N = EMB small -> split-K)
      packA(R.soft, softA, R.V);
      gemmSK(softA, R.embP, R.V, emb1, EMBD);
      packA(emb1, emb1A, EMBD);
      gemm(emb1A, R.g1P, EMBD, nullptr, nullptr, 0, R.g1b, y1, EMBD, 0);
      gemm(emb2A, R.g2P, EMBD, nullptr, nullptr, 0, R.g2b, y2, EMBD, 0);
      gate_mix_k<<<gblk(BATCH * EMBD), 256, 0, stream>>>(y1, y2, emb1, emb2,
                                                         R.embx, BATCH * EMBD);
      packA(R.embx, embxA, EMBD);
      xinA = embxA;
    }
    // LSTM layer 0: gates = x@Wih0^T + h_prev@Whh0^T + (bih0+bhh0)
    packA(R.h0b, h0A, HIDD);                            // old h0
    gemm(xinA, R.wih0, EMBD, h0A, R.whh0, HIDD, R.bsum0, gates, 4 * HIDD, 0);
    lstm_elem_k<<<gblk(nBH), 256, 0, stream>>>(gates, R.h0b, R.c0b);
    // LSTM layer 1
    packA(R.h0b, h0A, HIDD);                            // new h0
    packA(R.h1b, h1A, HIDD);                            // old h1
    gemm(h0A, R.wih1, HIDD, h1A, R.whh1, HIDD, R.bsum1, gates, 4 * HIDD, 0);
    lstm_elem_k<<<gblk(nBH), 256, 0, stream>>>(gates, R.h1b, R.c1b);
    // attention
    packA(R.h1b, h1A, HIDD);                            // new h1
    gemm(h1A, R.attinP, HIDD, nullptr, nullptr, 0, R.attinb, gamma, HIDD, 0);
    attn_scores_k<<<BATCH * SLEN, 256, 0, stream>>>(contexts, gamma, sc);
    softmax_rows_k<<<BATCH, 256, 0, stream>>>(sc, wts, SLEN);
    attn_ctx_k<<<BATCH, 256, 0, stream>>>(wts, contexts, ctxb);
    concat2_k<<<gblk(BATCH * 2 * HIDD), 256, 0, stream>>>(ctxb, R.h1b, cat);
    packA(cat, catA, 2 * HIDD);
    float* outT = R.outBase + (size_t)t * BATCH * HIDD;
    gemm(catA, R.attoutP, 2 * HIDD, nullptr, nullptr, 0, R.attoutb, outT, HIDD, 1);
    // vocab projection + softmax carry
    packA(outT, outA, HIDD);
    gemm(outA, R.linP, HIDD, nullptr, nullptr, 0, R.linb, logits, R.V, 0);
    softmax_rows_k<<<BATCH, 256, 0, stream>>>(logits, R.soft, R.V);
  };

  for (int t = 0; t < TSTEPS; ++t) {
    runStep(spk, t);
    runStep(dir, t);
    if (t >= 1)
      add2_k<<<gblk(BATCH * EMBD), 256, 0, stream>>>(
          embx, embxd, gemb + (size_t)(t - 1) * BATCH * EMBD, BATCH * EMBD);
  }
}
